// SplineCrossAttention_38465727103292
// MI455X (gfx1250) — compile-verified, matrix operand order checked by reading, not run on hardware
//
#include <hip/hip_runtime.h>

// ---------------------------------------------------------------------------
// SplineCrossAttention for MI455X (gfx1250, wave32, WMMA 16x16x32 f16/f32acc)
//
// Pipeline:
//   prep:    f32->f16 weight conversion + x transpose to [b, n, c]
//   proj:    q/k/v = W @ x   (WMMA, f16 in / f32 acc)
//   attn:    fused QK^T -> KAN spline -> online softmax -> attn@V,
//            flash-style, key dim split 2-ways per row tile, merged via LDS.
//            Spline evaluated as direct uniform cubic B-spline window
//            (4 bases instead of full Cox-de Boor recursion).
//   outproj: Wo @ o          (WMMA)
// ---------------------------------------------------------------------------

typedef _Float16 v16h __attribute__((ext_vector_type(16)));
typedef _Float16 v8h  __attribute__((ext_vector_type(8)));
typedef float    v8f  __attribute__((ext_vector_type(8)));

#define CH    256
#define INNER 128
#define HEADS 4
#define DH    32
#define NB    2
#define NPIX  1024
#define NCOEF 8

// Build a 16x16x32 WMMA A/B operand from two contiguous 8-half chunks.
// Operand layout (16-bit, wave32): lane L (g=L/16) holds K = 8g+t (t<8) and
// K = 16+8g+(t-8) (t>=8)  ->  two 16-byte loads per lane.
__device__ __forceinline__ v16h ld_ab(const _Float16* p0, const _Float16* p1) {
  v8h lo = *(const v8h*)p0;
  v8h hi = *(const v8h*)p1;
  v16h r;
#pragma unroll
  for (int t = 0; t < 8; ++t) { r[t] = lo[t]; r[t + 8] = hi[t]; }
  return r;
}

// Reductions across the 16-lane half-wave (xor masks 1,2,4,8 stay in-group).
__device__ __forceinline__ float red_max16(float v) {
  v = fmaxf(v, __shfl_xor(v, 1));
  v = fmaxf(v, __shfl_xor(v, 2));
  v = fmaxf(v, __shfl_xor(v, 4));
  v = fmaxf(v, __shfl_xor(v, 8));
  return v;
}
__device__ __forceinline__ float red_sum16(float v) {
  v += __shfl_xor(v, 1);
  v += __shfl_xor(v, 2);
  v += __shfl_xor(v, 4);
  v += __shfl_xor(v, 8);
  return v;
}

// KANLinear(1,1): base_w * silu(xs) + cubic B-spline on uniform knots.
// Knots t_j = -13.2 + 2.4*j. For xs in [-6,6] only 4 of the 8 bases are
// nonzero: window i = clamp(floor((xs+6)/2.4), 0, 4), local t in [0,1),
// weights = {(1-t)^3, 3t^3-6t^2+4, -3t^3+3t^2+3t+1, t^3} / 6.
// Identical to the Cox-de Boor reference (incl. clipped endpoint xs=6).
__device__ __forceinline__ float kan_eval(float v, float bw, const float* c) {
  float xs = (v != v) ? 0.0f : v;           // nan_to_num
  xs = fminf(fmaxf(xs, -6.0f), 6.0f);       // clip (handles +-inf too)
  float acc = bw * xs * __builtin_amdgcn_rcpf(1.0f + __expf(-xs));  // silu base

  float u = (xs + 6.0f) * (1.0f / 2.4f);    // u in [0, 5]
  int i = (int)u;                           // trunc == floor (u >= 0)
  i = (i > 4) ? 4 : i;
  float t = u - (float)i;
  float t2 = t * t, t3 = t2 * t, omt = 1.0f - t;
  float w0 = omt * omt * omt * (1.0f / 6.0f);
  float w1 = (3.0f * t3 - 6.0f * t2 + 4.0f) * (1.0f / 6.0f);
  float w2 = (-3.0f * t3 + 3.0f * t2 + 3.0f * t + 1.0f) * (1.0f / 6.0f);
  float w3 = t3 * (1.0f / 6.0f);

  // Branchless window select: (a0..a3) = c[i..i+3], i in [0,4]
  float a0 = c[0], a1 = c[1], a2 = c[2], a3 = c[3];
  a0 = (i > 0) ? c[1] : a0; a1 = (i > 0) ? c[2] : a1;
  a2 = (i > 0) ? c[3] : a2; a3 = (i > 0) ? c[4] : a3;
  a0 = (i > 1) ? c[2] : a0; a1 = (i > 1) ? c[3] : a1;
  a2 = (i > 1) ? c[4] : a2; a3 = (i > 1) ? c[5] : a3;
  a0 = (i > 2) ? c[3] : a0; a1 = (i > 2) ? c[4] : a1;
  a2 = (i > 2) ? c[5] : a2; a3 = (i > 2) ? c[6] : a3;
  a0 = (i > 3) ? c[4] : a0; a1 = (i > 3) ? c[5] : a1;
  a2 = (i > 3) ? c[6] : a2; a3 = (i > 3) ? c[7] : a3;

  acc += w0 * a0 + w1 * a1 + w2 * a2 + w3 * a3;
  return acc;
}

// ---------------------------------------------------------------------------
// prep: convert weights to f16, transpose x [b,c,n] -> xT [b,n,c]
// ---------------------------------------------------------------------------
__global__ void prep_kernel(const float* __restrict__ x,  const float* __restrict__ Wq,
                            const float* __restrict__ Wk, const float* __restrict__ Wv,
                            const float* __restrict__ Wo,
                            _Float16* __restrict__ xT,  _Float16* __restrict__ Wqh,
                            _Float16* __restrict__ Wkh, _Float16* __restrict__ Wvh,
                            _Float16* __restrict__ Woh) {
  const int stride = gridDim.x * blockDim.x;
  for (int idx = blockIdx.x * blockDim.x + threadIdx.x; idx < NB * CH * NPIX; idx += stride) {
    int n = idx & (NPIX - 1);
    int c = (idx >> 10) & (CH - 1);
    int b = idx >> 18;
    xT[((size_t)(b * NPIX) + n) * CH + c] = (_Float16)x[idx];
  }
  for (int idx = blockIdx.x * blockDim.x + threadIdx.x; idx < INNER * CH; idx += stride) {
    Wqh[idx] = (_Float16)Wq[idx];
    Wkh[idx] = (_Float16)Wk[idx];
    Wvh[idx] = (_Float16)Wv[idx];
    Woh[idx] = (_Float16)Wo[idx];
  }
}

// ---------------------------------------------------------------------------
// proj: y[i,n] = sum_c W[i,c] * x[b,c,n]  for Wq/Wk/Wv; one wave per 16x16 tile.
//   q,k stored as [b, head, n, d] (d contiguous)   -> A/B operands for QK^T
//   v   stored as [b, i, n]      (n contiguous)    -> B operand for attn@V
// ---------------------------------------------------------------------------
__global__ void proj_kernel(const _Float16* __restrict__ xT,
                            const _Float16* __restrict__ Wqh,
                            const _Float16* __restrict__ Wkh,
                            const _Float16* __restrict__ Wvh,
                            _Float16* __restrict__ qh, _Float16* __restrict__ kh,
                            _Float16* __restrict__ vh) {
  const int lane  = threadIdx.x & 31;
  const int w     = blockIdx.x * 8 + (threadIdx.x >> 5);
  const int itile = w & 7;
  const int ntile = (w >> 3) & 63;
  const int b     = (w >> 9) & 1;
  const int mat   = w >> 10;                  // 0=q, 1=k, 2=v
  const int g     = lane >> 4;
  const int l16   = lane & 15;

  const _Float16* Wh = (mat == 0) ? Wqh : (mat == 1) ? Wkh : Wvh;
  const _Float16* arow = Wh + (itile * 16 + l16) * CH;
  const _Float16* brow = xT + ((size_t)(b * NPIX) + ntile * 16 + l16) * CH;

  v8f acc = {};
#pragma unroll
  for (int kk = 0; kk < CH; kk += 32) {
    v16h a  = ld_ab(arow + kk + 8 * g, arow + kk + 16 + 8 * g);
    v16h bb = ld_ab(brow + kk + 8 * g, brow + kk + 16 + 8 * g);
    acc = __builtin_amdgcn_wmma_f32_16x16x32_f16(false, a, false, bb, (short)0, acc,
                                                 false, false);
  }
#pragma unroll
  for (int r = 0; r < 8; ++r) {
    int i = itile * 16 + 8 * g + r;           // D row
    int n = ntile * 16 + l16;                 // D col
    _Float16 hv = (_Float16)acc[r];
    if (mat == 2) {
      vh[((size_t)(b * INNER) + i) * NPIX + n] = hv;
    } else {
      int head = i >> 5, d = i & 31;
      _Float16* dst = (mat == 0) ? qh : kh;
      dst[(((size_t)(b * HEADS) + head) * NPIX + n) * DH + d] = hv;
    }
  }
}

// ---------------------------------------------------------------------------
// attn: fused QK^T -> KAN -> online softmax -> attn@V.
// Wave pairs split the 1024-key dim in half (512 keys each, chunks of 32:
// two 16x16 sim WMMAs -> one P A-operand -> two attn@V WMMAs), then merge
// their (m, s, O) state via LDS with the flash-attention combine.
// ---------------------------------------------------------------------------
__global__ void attn_kernel(const _Float16* __restrict__ qh,
                            const _Float16* __restrict__ kh,
                            const _Float16* __restrict__ vh,
                            _Float16* __restrict__ oh,
                            const float* __restrict__ base_w,
                            const float* __restrict__ spline_w,
                            const float* __restrict__ spline_scaler) {
  __shared__ __align__(16) _Float16 plds[8 * 512];  // 1KB P-staging per wave
  __shared__ __align__(16) float    olds[4 * 512];  // odd-half O tiles [16][32]
  __shared__ float                  slds[4 * 32];   // odd-half m[16], s[16]

  const int lane  = threadIdx.x & 31;
  const int wid   = threadIdx.x >> 5;
  const int w     = blockIdx.x * 8 + wid;
  const int pairg = w >> 1;                 // global row-tile id (512 total)
  const int half  = w & 1;                  // key-dim split id
  const int pairl = wid >> 1;               // local pair 0..3
  const int rtile = pairg & 63;
  const int h     = (pairg >> 6) & 3;
  const int b     = pairg >> 8;
  const int g     = lane >> 4;
  const int l16   = lane & 15;

  const float bw  = base_w[0];
  const float scl = spline_scaler[0];
  float coef[NCOEF];
#pragma unroll
  for (int j = 0; j < NCOEF; ++j) coef[j] = spline_w[j] * scl;

  const _Float16* qbase = qh + (size_t)(b * HEADS + h) * NPIX * DH;
  const _Float16* kbase = kh + (size_t)(b * HEADS + h) * NPIX * DH;
  const _Float16* vbase = vh + (size_t)(b * INNER + h * DH) * NPIX;

  // Q A-operand: row = rtile*16 + l16, K = full dh=32 (single WMMA K-step)
  const _Float16* qr = qbase + (rtile * 16 + l16) * DH;
  v16h qa = ld_ab(qr + 8 * g, qr + 16 + 8 * g);

  float m[8], s[8];
  v8f O0 = {}, O1 = {};
#pragma unroll
  for (int r = 0; r < 8; ++r) { m[r] = -__builtin_inff(); s[r] = 0.0f; }

  _Float16* P = plds + wid * 512;
  const float simscale = 0.17677669529663688f;   // 1/sqrt(32)
  const int jlo = half * (NPIX / 2);

  for (int jc = jlo; jc < jlo + NPIX / 2; jc += 32) {
    // K B-operands (k stored [.., key, d] -> K=d contiguous per lane)
    const _Float16* k0p = kbase + (jc + l16) * DH;
    const _Float16* k1p = kbase + (jc + 16 + l16) * DH;
    v16h kb0 = ld_ab(k0p + 8 * g, k0p + 16 + 8 * g);
    v16h kb1 = ld_ab(k1p + 8 * g, k1p + 16 + 8 * g);

    v8f z = {};
    v8f s0 = __builtin_amdgcn_wmma_f32_16x16x32_f16(false, qa, false, kb0, (short)0, z,
                                                    false, false);
    v8f s1 = __builtin_amdgcn_wmma_f32_16x16x32_f16(false, qa, false, kb1, (short)0, z,
                                                    false, false);

    float k0v[8], k1v[8];
#pragma unroll
    for (int r = 0; r < 8; ++r) {
      k0v[r] = kan_eval(s0[r] * simscale, bw, coef);
      k1v[r] = kan_eval(s1[r] * simscale, bw, coef);
    }

    // Online softmax update + stage P into LDS (D-layout -> row-major tile)
#pragma unroll
    for (int r = 0; r < 8; ++r) {
      float cm = red_max16(fmaxf(k0v[r], k1v[r]));
      float nm = fmaxf(m[r], cm);
      float rs = __expf(m[r] - nm);          // first iter: exp(-inf) = 0
      float p0 = __expf(k0v[r] - nm);
      float p1 = __expf(k1v[r] - nm);
      m[r] = nm;
      O0[r] *= rs;
      O1[r] *= rs;
      s[r] = s[r] * rs + red_sum16(p0 + p1);
      P[(8 * g + r) * 32 + l16]      = (_Float16)p0;
      P[(8 * g + r) * 32 + 16 + l16] = (_Float16)p1;
    }

    // Re-read P in A-operand layout (same-wave DS ops are in-order)
    const _Float16* pr = P + l16 * 32;
    v16h pa = ld_ab(pr + 8 * g, pr + 16 + 8 * g);

    // V B-operands (v stored [.., d, key] -> K=key contiguous per lane)
    const _Float16* v0p = vbase + (size_t)l16 * NPIX + jc;
    const _Float16* v1p = vbase + (size_t)(16 + l16) * NPIX + jc;
    v16h vb0 = ld_ab(v0p + 8 * g, v0p + 16 + 8 * g);
    v16h vb1 = ld_ab(v1p + 8 * g, v1p + 16 + 8 * g);

    O0 = __builtin_amdgcn_wmma_f32_16x16x32_f16(false, pa, false, vb0, (short)0, O0,
                                                false, false);
    O1 = __builtin_amdgcn_wmma_f32_16x16x32_f16(false, pa, false, vb1, (short)0, O1,
                                                false, false);
  }

  // ---- merge the two key-halves of each row tile ----
  if (half == 1) {
#pragma unroll
    for (int r = 0; r < 8; ++r) {
      olds[pairl * 512 + (8 * g + r) * 32 + l16]      = O0[r];
      olds[pairl * 512 + (8 * g + r) * 32 + 16 + l16] = O1[r];
    }
    if (l16 == 0) {
#pragma unroll
      for (int r = 0; r < 8; ++r) {
        slds[pairl * 32 + 8 * g + r]      = m[r];
        slds[pairl * 32 + 16 + 8 * g + r] = s[r];
      }
    }
  }
  __syncthreads();
  if (half == 0) {
#pragma unroll
    for (int r = 0; r < 8; ++r) {
      float pm = slds[pairl * 32 + 8 * g + r];
      float ps = slds[pairl * 32 + 16 + 8 * g + r];
      float nm = fmaxf(m[r], pm);
      float e0 = __expf(m[r] - nm);
      float e1 = __expf(pm - nm);
      float inv = 1.0f / (s[r] * e0 + ps * e1);
      float po0 = olds[pairl * 512 + (8 * g + r) * 32 + l16];
      float po1 = olds[pairl * 512 + (8 * g + r) * 32 + 16 + l16];
      float o0 = (O0[r] * e0 + po0 * e1) * inv;
      float o1 = (O1[r] * e0 + po1 * e1) * inv;
      int n = rtile * 16 + 8 * g + r;
      size_t base = ((size_t)(b * NPIX) + n) * INNER + h * DH;
      oh[base + l16]      = (_Float16)o0;
      oh[base + 16 + l16] = (_Float16)o1;
    }
  }
}

// ---------------------------------------------------------------------------
// outproj: out[b,c,n] = sum_i Wo[c,i] * o[b,n,i]; one wave per 16x16 tile.
// ---------------------------------------------------------------------------
__global__ void outproj_kernel(const _Float16* __restrict__ Woh,
                               const _Float16* __restrict__ oh,
                               float* __restrict__ out) {
  const int lane  = threadIdx.x & 31;
  const int w     = blockIdx.x * 8 + (threadIdx.x >> 5);
  const int ctile = w & 15;
  const int ntile = (w >> 4) & 63;
  const int b     = w >> 10;
  const int g     = lane >> 4;
  const int l16   = lane & 15;

  const _Float16* arow = Woh + (ctile * 16 + l16) * INNER;
  const _Float16* brow = oh + ((size_t)(b * NPIX) + ntile * 16 + l16) * INNER;

  v8f acc = {};
#pragma unroll
  for (int kk = 0; kk < INNER; kk += 32) {
    v16h a  = ld_ab(arow + kk + 8 * g, arow + kk + 16 + 8 * g);
    v16h bb = ld_ab(brow + kk + 8 * g, brow + kk + 16 + 8 * g);
    acc = __builtin_amdgcn_wmma_f32_16x16x32_f16(false, a, false, bb, (short)0, acc,
                                                 false, false);
  }
#pragma unroll
  for (int r = 0; r < 8; ++r) {
    int c = ctile * 16 + 8 * g + r;
    int n = ntile * 16 + l16;
    out[((size_t)(b * CH) + c) * NPIX + n] = acc[r];
  }
}

// ---------------------------------------------------------------------------
extern "C" void kernel_launch(void* const* d_in, const int* in_sizes, int n_in,
                              void* d_out, int out_size, void* d_ws, size_t ws_size,
                              hipStream_t stream) {
  (void)in_sizes; (void)n_in; (void)out_size; (void)ws_size;

  const float* x             = (const float*)d_in[0];
  const float* Wq            = (const float*)d_in[1];
  const float* Wk            = (const float*)d_in[2];
  const float* Wv            = (const float*)d_in[3];
  const float* Wo            = (const float*)d_in[4];
  const float* base_w        = (const float*)d_in[5];
  const float* spline_w      = (const float*)d_in[6];
  const float* spline_scaler = (const float*)d_in[7];
  float* out = (float*)d_out;

  // Workspace carving (all offsets 16B aligned): ~3.3 MB total
  _Float16* ws  = (_Float16*)d_ws;
  _Float16* xT  = ws;                                   // [B, N, CH]
  _Float16* Wqh = xT  + (size_t)NB * NPIX * CH;         // [INNER, CH]
  _Float16* Wkh = Wqh + (size_t)INNER * CH;
  _Float16* Wvh = Wkh + (size_t)INNER * CH;
  _Float16* Woh = Wvh + (size_t)INNER * CH;             // [CH, INNER]
  _Float16* qh  = Woh + (size_t)CH * INNER;             // [B, H, N, DH]
  _Float16* kh  = qh  + (size_t)NB * INNER * NPIX;      // [B, H, N, DH]
  _Float16* vh  = kh  + (size_t)NB * INNER * NPIX;      // [B, INNER, N]
  _Float16* oh  = vh  + (size_t)NB * INNER * NPIX;      // [B, N, INNER]

  prep_kernel<<<512, 256, 0, stream>>>(x, Wq, Wk, Wv, Wo, xT, Wqh, Wkh, Wvh, Woh);
  proj_kernel<<<384, 256, 0, stream>>>(xT, Wqh, Wkh, Wvh, qh, kh, vh);      // 3072 waves
  attn_kernel<<<128, 256, 0, stream>>>(qh, kh, vh, oh, base_w, spline_w,
                                       spline_scaler);                       // 1024 waves
  outproj_kernel<<<256, 256, 0, stream>>>(Woh, oh, out);                     // 2048 waves
}
